// GRU_N21_4724464025751
// MI455X (gfx1250) — matrix operand order repaired and checked by reading, not compiled, not run on hardware
//
#include <hip/hip_runtime.h>
#include <math.h>

typedef __attribute__((ext_vector_type(16))) __bf16 v16bf;
typedef __attribute__((ext_vector_type(8)))  float  v8f;

#define H_DIM 256
#define IN_DIM 256
#define NNB 8
#define W_ELEMS (H_DIM * H_DIM)   // 65536 elements per weight matrix

__device__ __forceinline__ float sigmoid_f(float x) { return 1.0f / (1.0f + __expf(-x)); }

__device__ __forceinline__ v8f wmma_bf16(v16bf a, v16bf b, v8f c) {
  // D = A(16x32 bf16) * B(32x16 bf16) + C(16x16 f32)
  return __builtin_amdgcn_wmma_f32_16x16x32_bf16(false, a, false, b, (short)0, c, false, false);
}

// A fragment (16x32) from f32 row-major [rows 0..15 of base], cols k0..k0+31.
// Layout: lane half 0 holds K = k0+0..15; half 1 holds K = k0+16..31 (matches B).
__device__ __forceinline__ v16bf frag_a_f32(const float* __restrict__ base, int ld, int k0, int lane) {
  const int m  = lane & 15;
  const int h16 = (lane >> 4) << 4;
  const float* r = base + (size_t)m * ld + k0;
  v16bf f;
#pragma unroll
  for (int i = 0; i < 16; ++i) f[i] = (__bf16)r[h16 + i];
  return f;
}

// Same pattern, but source is f32 in LDS.
__device__ __forceinline__ v16bf frag_a_ldsf32(const float* base, int ld, int k0, int lane) {
  const int m  = lane & 15;
  const int h16 = (lane >> 4) << 4;
  const float* r = base + m * ld + k0;
  v16bf f;
#pragma unroll
  for (int i = 0; i < 16; ++i) f[i] = (__bf16)r[h16 + i];
  return f;
}

// B fragment (32x16) from pre-converted bf16 weight, row-major [256,256].
// B(k,n) = W[n0+n, k0+k].  Lanes 0-15: K=k0+0..15 of column n; lanes 16-31: K=k0+16..31.
// 32 contiguous, 32B-aligned bytes per lane -> pure global_load_b128 x2, no VALU converts.
__device__ __forceinline__ v16bf frag_b_bf(const __bf16* __restrict__ Wbf, int n0, int k0, int lane) {
  const int n   = lane & 15;
  const int h16 = (lane >> 4) << 4;
  return *(const v16bf*)(Wbf + (size_t)(n0 + n) * H_DIM + k0 + h16);
}

// ---- one-time weight conversion f32 -> bf16 into workspace ----
__global__ __launch_bounds__(256) void convert_weights(
    const float* __restrict__ w0, const float* __restrict__ w1,
    const float* __restrict__ w2, const float* __restrict__ w3,
    const float* __restrict__ w4, const float* __restrict__ w5,
    __bf16* __restrict__ dst)
{
  const int i = blockIdx.x * blockDim.x + threadIdx.x;   // 0 .. 6*65536-1
  const int w = i >> 16;                                  // which matrix
  const int e = i & (W_ELEMS - 1);
  float v;
  if      (w == 0) v = w0[e];
  else if (w == 1) v = w1[e];
  else if (w == 2) v = w2[e];
  else if (w == 3) v = w3[e];
  else if (w == 4) v = w4[e];
  else             v = w5[e];
  dst[i] = (__bf16)v;
}

__global__ __launch_bounds__(32) void gru_fused(
    const float* __restrict__ x,   const float* __restrict__ h_sum,
    const float* __restrict__ hs,
    const __bf16* __restrict__ Wir, const float* __restrict__ bir,
    const __bf16* __restrict__ Whr, const float* __restrict__ bhr,
    const __bf16* __restrict__ Wiz, const float* __restrict__ biz,
    const __bf16* __restrict__ Whz, const float* __restrict__ bhz,
    const __bf16* __restrict__ Win, const float* __restrict__ bin_,
    const __bf16* __restrict__ Whn, const float* __restrict__ bhn,
    float* __restrict__ out, int Btot)
{
  __shared__ __bf16 lds_xr[16 * H_DIM];   // x@Wir^T + bir + bhr            (8 KB)
  __shared__ __bf16 lds_xn[16 * H_DIM];   // x@Win^T + bin + bhn            (8 KB)
  __shared__ __bf16 lds_z [16 * H_DIM];   // pre-z, then z = sigmoid(...)   (8 KB)
  __shared__ float  lds_s [16 * H_DIM];   // s accumulator (f32)            (16 KB)

  const int lane = threadIdx.x;          // wave32, one wave per workgroup
  const int nsub = lane & 15;            // column within 16-wide tile (C layout)
  const int crow = (lane >> 4) << 3;     // row offset for C layout (0 or 8)
  const int row0 = blockIdx.x * 16;      // this block's B-row strip

  const float* xb = x     + (size_t)row0 * IN_DIM;
  const float* hb = h_sum + (size_t)row0 * H_DIM;

  // -------- Phase 1a: input-side GEMMs from x fragments --------
  {
    v16bf xf[8];
#pragma unroll
    for (int kk = 0; kk < 8; ++kk) xf[kk] = frag_a_f32(xb, IN_DIM, kk * 32, lane);

    // overlap: prefetch first hs strip (neighbor 0) while doing the x GEMMs
    {
      const char* nxt = (const char*)(hs + (size_t)row0 * H_DIM);
#pragma unroll
      for (int p = 0; p < 4; ++p)
        __builtin_prefetch(nxt + (size_t)lane * 512 + p * 128, 0, 3);
    }

#pragma unroll 1
    for (int j = 0; j < 16; ++j) {
      const int c0 = j * 16, col = c0 + nsub;
      v8f aR = {}, aN = {}, aZ = {};
#pragma unroll
      for (int kk = 0; kk < 8; ++kk) {
        const int k0 = kk * 32;
        aR = wmma_bf16(xf[kk], frag_b_bf(Wir, c0, k0, lane), aR);
        aN = wmma_bf16(xf[kk], frag_b_bf(Win, c0, k0, lane), aN);
        aZ = wmma_bf16(xf[kk], frag_b_bf(Wiz, c0, k0, lane), aZ);
      }
      const float bR = bir[col] + bhr[col];
      const float bN = bin_[col] + bhn[col];
      const float bZ = biz[col] + bhz[col];
#pragma unroll
      for (int v = 0; v < 8; ++v) {
        const int rl = crow + v;
        lds_xr[rl * H_DIM + col] = (__bf16)(aR[v] + bR);
        lds_xn[rl * H_DIM + col] = (__bf16)(aN[v] + bN);
        lds_z [rl * H_DIM + col] = (__bf16)(aZ[v] + bZ);   // pre-z
      }
    }
  }

  // -------- Phase 1b: finish z with h_sum @ Whz^T, init s --------
  {
    v16bf hf[8];
#pragma unroll
    for (int kk = 0; kk < 8; ++kk) hf[kk] = frag_a_f32(hb, H_DIM, kk * 32, lane);

#pragma unroll 1
    for (int j = 0; j < 16; ++j) {
      const int c0 = j * 16, col = c0 + nsub;
      v8f aZ = {};
#pragma unroll
      for (int kk = 0; kk < 8; ++kk)
        aZ = wmma_bf16(hf[kk], frag_b_bf(Whz, c0, kk * 32, lane), aZ);
#pragma unroll
      for (int v = 0; v < 8; ++v) {
        const int rl = crow + v;
        const int idx = rl * H_DIM + col;
        lds_z[idx] = (__bf16)sigmoid_f(aZ[v] + (float)lds_z[idx]);
        lds_s[idx] = 0.0f;
      }
    }
  }

  // -------- Phase 2: neighbor loop — r_i = sigmoid(xr + hs_i@Whr^T), s += r_i * hs_i --------
#pragma unroll 1
  for (int nb = 0; nb < NNB; ++nb) {
    const float* hsb = hs + ((size_t)nb * Btot + row0) * H_DIM;
    v16bf af[8];
#pragma unroll
    for (int kk = 0; kk < 8; ++kk) af[kk] = frag_a_f32(hsb, H_DIM, kk * 32, lane);

    // prefetch next neighbor's 16KB strip (counter-free global_prefetch_b8)
    if (nb + 1 < NNB) {
      const char* nxt = (const char*)(hs + ((size_t)(nb + 1) * Btot + row0) * H_DIM);
#pragma unroll
      for (int p = 0; p < 4; ++p)
        __builtin_prefetch(nxt + (size_t)lane * 512 + p * 128, 0, 3);
    }

#pragma unroll 1
    for (int j = 0; j < 16; ++j) {
      const int c0 = j * 16, col = c0 + nsub;
      v8f acc = {};
#pragma unroll
      for (int kk = 0; kk < 8; ++kk)
        acc = wmma_bf16(af[kk], frag_b_bf(Whr, c0, kk * 32, lane), acc);
#pragma unroll
      for (int v = 0; v < 8; ++v) {
        const int rl = crow + v;
        const int idx = rl * H_DIM + col;
        const float r = sigmoid_f(acc[v] + (float)lds_xr[idx]);
        lds_s[idx] += r * hsb[(size_t)rl * H_DIM + col];   // same-wave LDS RMW: in-order
      }
    }
  }

  // make LDS s-tile writes visible before re-fragmenting (split dep counters on CDNA5)
  asm volatile("s_wait_dscnt 0" ::: "memory");

  // -------- Phase 3: n = tanh(xn + s@Whn^T), out = (1-z)*n + z*h_sum --------
  {
    v16bf sf[8];
#pragma unroll
    for (int kk = 0; kk < 8; ++kk) sf[kk] = frag_a_ldsf32(lds_s, H_DIM, kk * 32, lane);

    float* outb = out + (size_t)row0 * H_DIM;
#pragma unroll 1
    for (int j = 0; j < 16; ++j) {
      const int c0 = j * 16, col = c0 + nsub;
      v8f acc = {};
#pragma unroll
      for (int kk = 0; kk < 8; ++kk)
        acc = wmma_bf16(sf[kk], frag_b_bf(Whn, c0, kk * 32, lane), acc);
#pragma unroll
      for (int v = 0; v < 8; ++v) {
        const int rl = crow + v;
        const int idx = rl * H_DIM + col;
        const float nn = tanhf(acc[v] + (float)lds_xn[idx]);
        const float zz = (float)lds_z[idx];
        const float hh = hb[(size_t)rl * H_DIM + col];
        outb[(size_t)rl * H_DIM + col] = (1.0f - zz) * nn + zz * hh;
      }
    }
  }
}

extern "C" void kernel_launch(void* const* d_in, const int* in_sizes, int n_in,
                              void* d_out, int out_size, void* d_ws, size_t ws_size,
                              hipStream_t stream) {
  const float* x     = (const float*)d_in[0];
  const float* h_sum = (const float*)d_in[1];
  const float* hs    = (const float*)d_in[2];
  const float* Wir   = (const float*)d_in[3];
  const float* bir   = (const float*)d_in[4];
  const float* Whr   = (const float*)d_in[5];
  const float* bhr   = (const float*)d_in[6];
  const float* Wiz   = (const float*)d_in[7];
  const float* biz   = (const float*)d_in[8];
  const float* Whz   = (const float*)d_in[9];
  const float* bhz   = (const float*)d_in[10];
  const float* Win   = (const float*)d_in[11];
  const float* bin_  = (const float*)d_in[12];
  const float* Whn   = (const float*)d_in[13];
  const float* bhn   = (const float*)d_in[14];

  // bf16 weight cache in workspace: [Wir | Whr | Wiz | Whz | Win | Whn]
  __bf16* wbf = (__bf16*)d_ws;                 // needs 6 * 65536 * 2 = 768 KB
  const __bf16* bWir = wbf + 0 * W_ELEMS;
  const __bf16* bWhr = wbf + 1 * W_ELEMS;
  const __bf16* bWiz = wbf + 2 * W_ELEMS;
  const __bf16* bWhz = wbf + 3 * W_ELEMS;
  const __bf16* bWin = wbf + 4 * W_ELEMS;
  const __bf16* bWhn = wbf + 5 * W_ELEMS;

  convert_weights<<<dim3(6 * W_ELEMS / 256), dim3(256), 0, stream>>>(
      Wir, Whr, Wiz, Whz, Win, Whn, wbf);

  const int Btot = in_sizes[0] / IN_DIM;   // 32768
  dim3 grid(Btot / 16);
  dim3 block(32);
  gru_fused<<<grid, block, 0, stream>>>(x, h_sum, hs,
                                        bWir, bir, bWhr, bhr,
                                        bWiz, biz, bWhz, bhz,
                                        bWin, bin_, bWhn, bhn,
                                        (float*)d_out, Btot);
}
